// Q8Linear_19155554140371
// MI455X (gfx1250) — 8-bit matrix operand order repaired and checked by reading, compiled, not run on hardware
//
#include <hip/hip_runtime.h>
#include <cstdint>


typedef __attribute__((ext_vector_type(2))) int v2i;
typedef __attribute__((ext_vector_type(4))) int v4i;
typedef __attribute__((ext_vector_type(8))) int v8i;

constexpr int Mdim = 8192;
constexpr int Kdim = 4096;
constexpr int Ndim = 4096;

constexpr int BM = 128;
constexpr int BN = 256;
constexpr int BK = 64;
constexpr int NK = Kdim / BK;

constexpr int A_PITCH = 80;
constexpr int B_PITCH = 80;
constexpr int A_BYTES = BM * A_PITCH;
constexpr int B_BYTES = BN * B_PITCH;
constexpr int STAGE_BYTES = A_BYTES + B_BYTES;

__device__ __forceinline__ uint32_t lds_addr_u32(const void* p) {
  return (uint32_t)(uintptr_t)p;
}

__device__ __forceinline__ void async_ld_b128(uint32_t lds_byte_addr,
                                              uint32_t global_off,
                                              const void* sbase) {
  asm volatile("global_load_async_to_lds_b128 %0, %1, %2"
               :
               : "v"(lds_byte_addr), "v"(global_off), "s"(sbase)
               : "memory");
}

__global__ __launch_bounds__(256) void q8_gemm_wmma_gfx1250(
    const int8_t* __restrict__ x,
    const int8_t* __restrict__ w,
    const float* __restrict__ bias,
    const float* __restrict__ xsc,
    const float* __restrict__ wsc,
    const int* __restrict__ fuse_p,
    float* __restrict__ out)
{
  __shared__ __align__(16) char lds[2][STAGE_BYTES];

  const int tid  = threadIdx.x;
  const int lane = tid & 31;
  const int wid  = tid >> 5;
  const int wm   = wid >> 2;
  const int wn   = wid & 3;

  const int blockM = blockIdx.y * BM;
  const int blockN = blockIdx.x * BN;

  auto stage = [&](int s, int kt) {
    const uint32_t aB = lds_addr_u32(&lds[s][0]);
    const uint32_t bB = lds_addr_u32(&lds[s][A_BYTES]);
#pragma unroll
    for (int i = 0; i < 2; ++i) {
      int c   = tid + i * 256;
      int row = c >> 2, k16 = c & 3;
      async_ld_b128(aB + row * A_PITCH + k16 * 16,
                    (uint32_t)((blockM + row) * Kdim + kt * BK + k16 * 16), x);
    }
#pragma unroll
    for (int i = 0; i < 4; ++i) {
      int c   = tid + i * 256;
      int row = c >> 2, k16 = c & 3;
      async_ld_b128(bB + row * B_PITCH + k16 * 16,
                    (uint32_t)((blockN + row) * Kdim + kt * BK + k16 * 16), w);
    }
  };

  v8i acc[4][4];
  const v8i vzero = {0, 0, 0, 0, 0, 0, 0, 0};
#pragma unroll
  for (int i = 0; i < 4; ++i)
#pragma unroll
    for (int j = 0; j < 4; ++j) acc[i][j] = vzero;

  stage(0, 0);

  const int mloc = lane & 15;
  const int half = lane >> 4;

  for (int kt = 0; kt < NK; ++kt) {
    const int cur = kt & 1;
    if (kt + 1 < NK) {
      stage(cur ^ 1, kt + 1);
      asm volatile("s_wait_asynccnt 6" ::: "memory");
    } else {
      asm volatile("s_wait_asynccnt 0" ::: "memory");
    }
    __syncthreads();

    const char* Ab = &lds[cur][0];
    const char* Bb = &lds[cur][A_BYTES];

    v8i bf[4];
#pragma unroll
    for (int j = 0; j < 4; ++j) {
      const char* rp = Bb + (wn * 64 + j * 16 + mloc) * B_PITCH + half * 16;
      v4i q0 = *(const v4i*)(rp + 0);
      v4i q1 = *(const v4i*)(rp + 32);
      v8i f;
      f[0] = q0[0]; f[1] = q0[1]; f[2] = q0[2]; f[3] = q0[3];
      f[4] = q1[0]; f[5] = q1[1]; f[6] = q1[2]; f[7] = q1[3];
      bf[j] = f;
    }

#pragma unroll
    for (int i = 0; i < 4; ++i) {
      const char* rp = Ab + (wm * 64 + i * 16 + mloc) * A_PITCH + half * 16;
      v2i p0 = *(const v2i*)(rp + 0);
      v2i p1 = *(const v2i*)(rp + 8);
      v2i p2 = *(const v2i*)(rp + 32);
      v2i p3 = *(const v2i*)(rp + 40);
      v8i a;
      a[0] = p0[0]; a[1] = p0[1]; a[2] = p1[0]; a[3] = p1[1];
      a[4] = p2[0]; a[5] = p2[1]; a[6] = p3[0]; a[7] = p3[1];
#pragma unroll
      for (int j = 0; j < 4; ++j)
        acc[i][j] = __builtin_amdgcn_wmma_i32_16x16x64_iu8(
             true, a,  true, bf[j], acc[i][j],
             false,  false);
    }

    __syncthreads();
  }

  const int fg    = *fuse_p;
  const int ncol  = lane & 15;
  const int lhalf = lane >> 4;

  float wj[4], bj[4];
#pragma unroll
  for (int j = 0; j < 4; ++j) {
    const int gn = blockN + wn * 64 + j * 16 + ncol;
    wj[j] = wsc[gn];
    bj[j] = bias[gn];
  }

#pragma unroll
  for (int i = 0; i < 4; ++i) {
    const int gm0 = blockM + wm * 64 + i * 16 + lhalf * 8;
#pragma unroll
    for (int r = 0; r < 8; ++r) {
      const int gm = gm0 + r;
      const float xs = xsc[gm];
      float* orow = out + (size_t)gm * Ndim + (blockN + wn * 64 + ncol);
#pragma unroll
      for (int j = 0; j < 4; ++j) {
        float v = fmaf((float)acc[i][j][r], xs * wj[j], bj[j]);
        if (fg) {
          const float u = v * fmaf(0.044715f * v, v, 1.0f);
          const float e = __builtin_amdgcn_exp2f(-2.3022082f * u);
          v = v * __builtin_amdgcn_rcpf(1.0f + e);
        }
        orow[j * 16] = v;
      }
    }
  }
}

extern "C" void kernel_launch(void* const* d_in, const int* in_sizes, int n_in,
                              void* d_out, int out_size, void* d_ws,
                              size_t ws_size, hipStream_t stream) {
  const int8_t* x    = (const int8_t*)d_in[0];
  const int8_t* w    = (const int8_t*)d_in[1];
  const float*  bias = (const float*)d_in[2];
  const float*  xsc  = (const float*)d_in[3];
  const float*  wsc  = (const float*)d_in[4];
  const int*    fuse = (const int*)d_in[5];
  float* out = (float*)d_out;

  dim3 grid(Ndim / BN, Mdim / BM);
  q8_gemm_wmma_gfx1250<<<grid, dim3(256), 0, stream>>>(x, w, bias, xsc, wsc,
                                                       fuse, out);
}
